// MyMultiAttention_34351148434040
// MI455X (gfx1250) — matrix operand order repaired and checked by reading, not compiled, not run on hardware
//
#include <hip/hip_runtime.h>
#include <hip/hip_bf16.h>

typedef __bf16 bf16;
typedef __attribute__((ext_vector_type(16))) __bf16 v16bf;
typedef __attribute__((ext_vector_type(8)))  __bf16 v8bf;
typedef __attribute__((ext_vector_type(8)))  float   v8f;
typedef __attribute__((ext_vector_type(4)))  int     v4i;

#define WMMA_BF16(a, b, c) \
  __builtin_amdgcn_wmma_f32_16x16x32_bf16(false, (a), false, (b), (short)0, (c), false, false)

constexpr int Bb = 4, Ss = 2048, Ee = 512, Hh = 8, Dd = 64, HD = 512;
constexpr float NEGBIG = -1.0e13f;

// ---------------------------------------------------------------------------
// CDNA5 async global->LDS copy (ASYNCcnt path); guarded fallback to sync copy.
#if defined(__has_builtin)
#if __has_builtin(__builtin_amdgcn_global_load_async_to_lds_b128)
#define HAVE_ASYNC_LDS 1
#endif
#endif

typedef __attribute__((address_space(1))) v4i* as1_v4i_ptr;
typedef __attribute__((address_space(3))) v4i* as3_v4i_ptr;

__device__ inline void cp16_to_lds(bf16* lds, const bf16* g) {
#if defined(HAVE_ASYNC_LDS)
  __builtin_amdgcn_global_load_async_to_lds_b128(
      (as1_v4i_ptr)(void*)g, (as3_v4i_ptr)(void*)lds, 0, 0);
#else
  *(v8bf*)lds = *(const v8bf*)g;
#endif
}

__device__ inline void wait_lds_stage() {
#if defined(HAVE_ASYNC_LDS)
#if __has_builtin(__builtin_amdgcn_s_wait_asynccnt)
  __builtin_amdgcn_s_wait_asynccnt(0);
#else
  asm volatile("s_wait_asynccnt 0" ::: "memory");
#endif
#endif
  __syncthreads();
}

// ---------------------------------------------------------------------------
// Fragment loaders (wave32 16-bit; contiguous k order shared by A and B, not the CDNA5 ISA §7.12.2 table order)
// A fragment 16x32 (MxK): lane L -> M = L%16; element e -> K = k0 + 16*hi + e
__device__ inline v16bf load_a_frag(const bf16* base, int row, int ld, int k0, int lane) {
  int hi = (lane >> 4) & 1;
  const bf16* p = base + (size_t)row * ld + k0 + hi * 16;
  v8bf lo = *(const v8bf*)p;
  v8bf up = *(const v8bf*)(p + 8);
  v16bf f;
#pragma unroll
  for (int e = 0; e < 8; ++e) { f[e] = lo[e]; f[e + 8] = up[e]; }
  return f;
}

// B fragment 32x16 (KxN): lane L -> N = L%16; element e -> K = k0 + 16*hi + e
__device__ inline v16bf load_b_frag(const bf16* base, int rowN, int ld, int k0, int lane) {
  int hi = (lane >> 4) & 1;
  const bf16* p = base + (size_t)rowN * ld + k0 + hi * 16;
  v8bf lo = *(const v8bf*)p;
  v8bf up = *(const v8bf*)(p + 8);
  v16bf f;
#pragma unroll
  for (int e = 0; e < 8; ++e) { f[e] = lo[e]; f[e + 8] = up[e]; }
  return f;
}

// ---------------------------------------------------------------------------
__global__ void cvt_f32_bf16(const float* __restrict__ in, bf16* __restrict__ out, int n) {
  int i = blockIdx.x * blockDim.x + threadIdx.x;
  if (i < n) out[i] = (bf16)in[i];
}

__global__ void transpose_cvt(const float* __restrict__ in, bf16* __restrict__ out, int R, int C) {
  int i = blockIdx.x * blockDim.x + threadIdx.x;
  if (i < R * C) {
    int r = i / C, c = i - r * C;
    out[(size_t)c * R + r] = (bf16)in[i];
  }
}

// ---------------------------------------------------------------------------
// GEMM: Y[M][N] = A[M][K] @ Bt[N][K]^T + bias[N],  K = 512, N tile = 64/block.
// Bt tile (64x512) staged once per block in LDS (async), shared by 8 waves.
// Each wave: 32 rows x 64 cols = 8 accumulators, 8 WMMA per 32-k step.
// mode 0: bf16 scatter into [B,H,S,D]; mode 1: f32 row-major [M][N].
constexpr int BLDS_LD = 520;   // 512 + 8 pad -> 1040B row stride, conflict-free
__global__ __launch_bounds__(256) void gemm_bf16_wmma(
    const bf16* __restrict__ A, const bf16* __restrict__ Bt,
    const float* __restrict__ bias,
    bf16* __restrict__ out_bf, float* __restrict__ out_f,
    int M, int N, int K, int mode) {
  __shared__ bf16 Blds[64 * BLDS_LD];

  int tid = threadIdx.x;
  int wave = tid >> 5;
  int lane = tid & 31;
  int col16 = lane & 15;
  int hi = (lane >> 4) & 1;

  int m0 = blockIdx.x * 256 + wave * 32;
  int n0 = blockIdx.y * 64;

  // stage Bt tile: 64 rows x 512 elems = 4096 x 16B chunks, 16 per thread
  for (int c = tid; c < 4096; c += 256) {
    int n = c >> 6, ch = (c & 63) * 8;
    cp16_to_lds(&Blds[n * BLDS_LD + ch], Bt + (size_t)(n0 + n) * K + ch);
  }
  wait_lds_stage();

  int arow0 = m0 + col16;
  int arow1 = m0 + 16 + col16;
  v8f acc[2][4] = {{v8f{}, v8f{}, v8f{}, v8f{}}, {v8f{}, v8f{}, v8f{}, v8f{}}};

  for (int k0 = 0; k0 < K; k0 += 32) {
    __builtin_prefetch(A + (size_t)arow0 * K + k0 + 64, 0, 1);
    v16bf a0 = load_a_frag(A, arow0, K, k0, lane);
    v16bf a1 = load_a_frag(A, arow1, K, k0, lane);
#pragma unroll
    for (int nt = 0; nt < 4; ++nt) {
      v16bf b = load_b_frag(&Blds[0], nt * 16 + col16, BLDS_LD, k0, lane);
      acc[0][nt] = WMMA_BF16(a0, b, acc[0][nt]);
      acc[1][nt] = WMMA_BF16(a1, b, acc[1][nt]);
    }
  }

#pragma unroll
  for (int rg = 0; rg < 2; ++rg)
#pragma unroll
    for (int nt = 0; nt < 4; ++nt) {
      int col = n0 + nt * 16 + col16;        // C layout: N = lane%16
      float bv = bias[col];
#pragma unroll
      for (int r = 0; r < 8; ++r) {          // C layout: M = r + 8*hi
        int grow = m0 + rg * 16 + r + 8 * hi;
        float v = acc[rg][nt][r] + bv;
        if (mode == 0) {
          int h = col >> 6, d = col & 63;
          int bi = grow >> 11, s = grow & (Ss - 1);
          out_bf[(((size_t)bi * Hh + h) * Ss + s) * Dd + d] = (bf16)v;
        } else {
          out_f[(size_t)grow * N + col] = v;
        }
      }
    }
}

// ---------------------------------------------------------------------------
// Flash attention: 256 threads = 8 waves; block covers 128 query rows of one
// (b,h). K/V 32x64 tiles staged in LDS once per key block, shared by 8 waves.
constexpr int KV_LD = 72;   // 64 + 8 pad -> 144B stride, 16B aligned, conflict-free
constexpr int P_LD = 40;    // 32 + 8 pad -> 80B stride
__global__ __launch_bounds__(256) void attn_wmma(
    const bf16* __restrict__ qb, const bf16* __restrict__ kb,
    const bf16* __restrict__ vb, const long long* __restrict__ pad,
    bf16* __restrict__ att_out) {
  __shared__ bf16 Klds[32 * KV_LD];
  __shared__ bf16 Vlds[32 * KV_LD];
  __shared__ bf16 Plds[8][16 * P_LD];

  int tid = threadIdx.x;
  int wave = tid >> 5;
  int lane = tid & 31;
  int col16 = lane & 15;
  int hi = (lane >> 4) & 1;

  int qblk = blockIdx.x & 15;          // S/128 = 16 tiles
  int bh = blockIdx.x >> 4;
  int h = bh & (Hh - 1);
  int b = bh >> 3;

  const bf16* Q = qb + (size_t)bh * Ss * Dd;
  const bf16* Km = kb + (size_t)bh * Ss * Dd;
  const bf16* Vm = vb + (size_t)bh * Ss * Dd;
  const long long* padb = pad + (size_t)b * Ss;
  int q0 = qblk * 128 + wave * 16;

  v16bf qa0 = load_a_frag(Q, q0 + col16, Dd, 0, lane);
  v16bf qa1 = load_a_frag(Q, q0 + col16, Dd, 32, lane);

  bool rowpad[8];
#pragma unroll
  for (int r = 0; r < 8; ++r) rowpad[r] = (padb[q0 + r + 8 * hi] == 1);

  float m_i[8], l_i[8];
  v8f o[4] = {v8f{}, v8f{}, v8f{}, v8f{}};
#pragma unroll
  for (int r = 0; r < 8; ++r) { m_i[r] = -3.0e38f; l_i[r] = 0.0f; }

  bf16* Pw = &Plds[wave][0];
  int srow = tid >> 3, sch = (tid & 7) * 8;   // staging: 256 threads = 32x8 chunks

  for (int j0 = 0; j0 < Ss; j0 += 32) {
    // cooperative stage of K and V tiles (async -> ASYNCcnt)
    cp16_to_lds(&Klds[srow * KV_LD + sch], Km + (size_t)(j0 + srow) * Dd + sch);
    cp16_to_lds(&Vlds[srow * KV_LD + sch], Vm + (size_t)(j0 + srow) * Dd + sch);
    wait_lds_stage();

    // scores: two 16x16 tiles from LDS K fragments
    v8f c0 = {}, c1 = {};
    {
      v16bf kb0 = load_b_frag(&Klds[0], col16, KV_LD, 0, lane);
      v16bf kb1 = load_b_frag(&Klds[0], col16, KV_LD, 32, lane);
      c0 = WMMA_BF16(qa0, kb0, c0);
      c0 = WMMA_BF16(qa1, kb1, c0);
      v16bf kb2 = load_b_frag(&Klds[0], 16 + col16, KV_LD, 0, lane);
      v16bf kb3 = load_b_frag(&Klds[0], 16 + col16, KV_LD, 32, lane);
      c1 = WMMA_BF16(qa0, kb2, c1);
      c1 = WMMA_BF16(qa1, kb3, c1);
    }

    bool cpad0 = (padb[j0 + col16] == 1);
    bool cpad1 = (padb[j0 + 16 + col16] == 1);
    float s0[8], s1[8];
#pragma unroll
    for (int r = 0; r < 8; ++r) {
      s0[r] = (cpad0 || rowpad[r]) ? NEGBIG : c0[r] * 0.125f;
      s1[r] = (cpad1 || rowpad[r]) ? NEGBIG : c1[r] * 0.125f;
    }

    // row reductions across half-wave (xor 1,2,4,8 stays within 16 lanes)
    float bm[8];
#pragma unroll
    for (int r = 0; r < 8; ++r) bm[r] = fmaxf(s0[r], s1[r]);
#pragma unroll
    for (int off = 1; off < 16; off <<= 1)
#pragma unroll
      for (int r = 0; r < 8; ++r) bm[r] = fmaxf(bm[r], __shfl_xor(bm[r], off));

    float f[8], ps0[8], ps1[8], rs[8];
#pragma unroll
    for (int r = 0; r < 8; ++r) {
      float nm = fmaxf(m_i[r], bm[r]);
      f[r] = __expf(m_i[r] - nm);
      m_i[r] = nm;
      ps0[r] = __expf(s0[r] - nm);
      ps1[r] = __expf(s1[r] - nm);
      rs[r] = ps0[r] + ps1[r];
    }
#pragma unroll
    for (int off = 1; off < 16; off <<= 1)
#pragma unroll
      for (int r = 0; r < 8; ++r) rs[r] += __shfl_xor(rs[r], off);
#pragma unroll
    for (int r = 0; r < 8; ++r) l_i[r] = l_i[r] * f[r] + rs[r];

#pragma unroll
    for (int dt = 0; dt < 4; ++dt)
#pragma unroll
      for (int r = 0; r < 8; ++r) o[dt][r] *= f[r];

    // C-layout -> A-fragment round trip through per-wave LDS slice
#pragma unroll
    for (int r = 0; r < 8; ++r) {
      Pw[(r + 8 * hi) * P_LD + col16] = (bf16)ps0[r];
      Pw[(r + 8 * hi) * P_LD + col16 + 16] = (bf16)ps1[r];
    }
    asm volatile("s_wait_dscnt 0" ::: "memory");

    v16bf pa = load_a_frag(Pw, col16, P_LD, 0, lane);

    // V B-fragments from LDS: B[k=j][n=d]; element e -> j = 16*hi + e
#pragma unroll
    for (int dt = 0; dt < 4; ++dt) {
      int d = dt * 16 + col16;
      v16bf vf;
#pragma unroll
      for (int e = 0; e < 16; ++e) vf[e] = Vlds[(16 * hi + e) * KV_LD + d];
      o[dt] = WMMA_BF16(pa, vf, o[dt]);
    }
    __syncthreads();   // protect K/V tiles before next stage overwrites them
  }

  float inv[8];
#pragma unroll
  for (int r = 0; r < 8; ++r) inv[r] = 1.0f / l_i[r];
#pragma unroll
  for (int dt = 0; dt < 4; ++dt)
#pragma unroll
    for (int r = 0; r < 8; ++r) {
      int sr = q0 + r + 8 * hi;
      int col = h * 64 + dt * 16 + col16;
      att_out[(size_t)(b * Ss + sr) * HD + col] = (bf16)(o[dt][r] * inv[r]);
    }
}

// ---------------------------------------------------------------------------
extern "C" void kernel_launch(void* const* d_in, const int* in_sizes, int n_in,
                              void* d_out, int out_size, void* d_ws, size_t ws_size,
                              hipStream_t stream) {
  const float* x = (const float*)d_in[0];
  const long long* pad = (const long long*)d_in[1];
  const float* Wq = (const float*)d_in[2];
  const float* bq = (const float*)d_in[3];
  const float* Wk = (const float*)d_in[4];
  const float* bk = (const float*)d_in[5];
  const float* Wv = (const float*)d_in[6];
  const float* bv = (const float*)d_in[7];
  const float* Wo = (const float*)d_in[8];
  const float* bo = (const float*)d_in[9];
  float* out = (float*)d_out;

  const size_t M = (size_t)Bb * Ss;          // 8192
  char* ws = (char*)d_ws;
  size_t off = 0;
  bf16* xb  = (bf16*)(ws + off); off += M * Ee * 2;
  bf16* WqT = (bf16*)(ws + off); off += (size_t)Ee * HD * 2;
  bf16* WkT = (bf16*)(ws + off); off += (size_t)Ee * HD * 2;
  bf16* WvT = (bf16*)(ws + off); off += (size_t)Ee * HD * 2;
  bf16* WoT = (bf16*)(ws + off); off += (size_t)HD * Ee * 2;
  bf16* qbuf = (bf16*)(ws + off); off += M * HD * 2;         // [B,H,S,D]
  bf16* kbuf = (bf16*)(ws + off); off += M * HD * 2;
  bf16* vbuf = (bf16*)(ws + off); off += M * HD * 2;
  bf16* attn_o = (bf16*)(ws + off); off += M * HD * 2;       // [B*S][H*D]

  {
    int n = (int)(M * Ee);
    cvt_f32_bf16<<<(n + 255) / 256, 256, 0, stream>>>(x, xb, n);
    int nw = Ee * HD;
    transpose_cvt<<<(nw + 255) / 256, 256, 0, stream>>>(Wq, WqT, Ee, HD);
    transpose_cvt<<<(nw + 255) / 256, 256, 0, stream>>>(Wk, WkT, Ee, HD);
    transpose_cvt<<<(nw + 255) / 256, 256, 0, stream>>>(Wv, WvT, Ee, HD);
    transpose_cvt<<<(nw + 255) / 256, 256, 0, stream>>>(Wo, WoT, HD, Ee);
  }

  {
    dim3 g((unsigned)(M / 256), HD / 64);
    gemm_bf16_wmma<<<g, 256, 0, stream>>>(xb, WqT, bq, qbuf, nullptr, (int)M, HD, Ee, 0);
    gemm_bf16_wmma<<<g, 256, 0, stream>>>(xb, WkT, bk, kbuf, nullptr, (int)M, HD, Ee, 0);
    gemm_bf16_wmma<<<g, 256, 0, stream>>>(xb, WvT, bv, vbuf, nullptr, (int)M, HD, Ee, 0);
  }

  {
    int nblk = Bb * Hh * (Ss / 128);         // 512 blocks x 256 threads
    attn_wmma<<<nblk, 256, 0, stream>>>(qbuf, kbuf, vbuf, pad, attn_o);
  }

  {
    dim3 g((unsigned)(M / 256), Ee / 64);
    gemm_bf16_wmma<<<g, 256, 0, stream>>>(attn_o, WoT, bo, nullptr, out, (int)M, Ee, Ee, 1);
  }
}
